// CandlestickLSTM_6012954214432
// MI455X (gfx1250) — matrix operand order repaired and checked by reading, not compiled, not run on hardware
//
#include <hip/hip_runtime.h>

// ---------------------------------------------------------------------------
// Fused seq2seq LSTM for MI455X (gfx1250), wave32 + WMMA bf16.
// One persistent kernel: each block owns 32 batch rows and runs all 64
// encoder + 32 decoder steps. Weights pre-converted fp32->bf16 into d_ws.
//  - Weight pointers laundered per timestep (blocks LICM -> scratch spills),
//    pinned to address_space(1) => global_load_b128 (LOADcnt only).
//  - B-fragment stream software-pipelined with a rotating 3-buffer and
//    prefetch distance 2 so two global loads stay in flight per wave.
//  - Cell states c0/c1 live in LDS (frees 32 VGPRs for the pipeline).
// ---------------------------------------------------------------------------

#define SEQ      64
#define TLEN     32
#define HDIM     256
#define MROWS    32        // batch rows per workgroup (2 WMMA M-tiles)
#define NTHREADS 512       // 16 waves of 32
#define XROW     (SEQ * 4) // floats per batch row of x

#define GLB __attribute__((address_space(1)))

typedef __attribute__((ext_vector_type(16))) __bf16 bf16x16;
typedef __attribute__((ext_vector_type(8)))  __bf16 bf16x8;
typedef __attribute__((ext_vector_type(8)))  float  f32x8;

union Frag16 { bf16x16 v; bf16x8 p[2]; };

// Launder a uniform value through an SGPR pair: blocks LICM/CSE across
// timesteps while costing nothing at runtime.
template <typename T>
__device__ __forceinline__ T opaque(T p) {
  asm volatile("" : "+s"(p));
  return p;
}

__device__ __forceinline__ __bf16 f2bf(float f) {
  union { float f; unsigned u; } x; x.f = f;
  unsigned r = (x.u + 0x7FFFu + ((x.u >> 16) & 1u)) >> 16; // RNE
  union { unsigned short s; __bf16 b; } y; y.s = (unsigned short)r;
  return y.b;
}
__device__ __forceinline__ float bf2f(__bf16 b) {
  union { unsigned short s; __bf16 b; } y; y.b = b;
  union { unsigned u; float f; } x; x.u = ((unsigned)y.s) << 16;
  return x.f;
}
__device__ __forceinline__ float sigf(float x)      { return 1.0f / (1.0f + __expf(-x)); }
__device__ __forceinline__ float tanhfast(float x)  { return 1.0f - 2.0f / (__expf(2.0f * x) + 1.0f); }
__device__ __forceinline__ f32x8 vsplat(float x) {
  f32x8 v;
#pragma unroll
  for (int i = 0; i < 8; ++i) v[i] = x;
  return v;
}

__device__ __forceinline__ void loadA(Frag16 (&a)[2], const __bf16* As,
                                      int nn, int hs, int kb) {
#pragma unroll
  for (int mt = 0; mt < 2; ++mt) {
    const __bf16* ab = As + (mt * 16 + nn) * HDIM + kb * 32 + hs * 8;
    a[mt].p[0] = *(const bf16x8*)(ab);        // K {0..7} / {8..15}
    a[mt].p[1] = *(const bf16x8*)(ab + 16);   // K {16..23} / {24..31}
  }
}

__device__ __forceinline__ void loadB(Frag16& b, const GLB __bf16* Wbase, int i) {
  const int kb = i >> 2, g = i & 3;
  const GLB __bf16* wb = Wbase + g * 256 * HDIM + kb * 32;
  b.p[0] = *(const GLB bf16x8*)(wb);
  b.p[1] = *(const GLB bf16x8*)(wb + 16);
}

// acc[mt][gate] += h_tile @ W^T tile. Per-wave hidden block `hb` (16 units),
// all 4 gates. A from LDS (row-major bf16 [32][256]); B streamed from global
// with a rotating 3-buffer, prefetch distance 2 (two loads in flight).
__device__ __forceinline__ void gemm_acc(f32x8 (&acc)[2][4],
                                         const __bf16* __restrict__ As,
                                         const GLB __bf16* __restrict__ W,
                                         int lane, int hb) {
  const int nn = lane & 15, hs = lane >> 4;
  const GLB __bf16* Wbase = W + (hb * 16 + nn) * HDIM + hs * 8;

  Frag16 a[2];
  Frag16 bb[3];
  loadA(a, As, nn, hs, 0);
  loadB(bb[0], Wbase, 0);
  loadB(bb[1], Wbase, 1);
#pragma unroll
  for (int i = 0; i < 32; ++i) {
    if (i + 2 < 32) loadB(bb[(i + 2) % 3], Wbase, i + 2);
    if ((i & 3) == 0 && i != 0) loadA(a, As, nn, hs, i >> 2);
    const int g = i & 3;
    acc[0][g] = __builtin_amdgcn_wmma_f32_16x16x32_bf16(
        false, a[0].v, false, bb[i % 3].v, (short)0, acc[0][g], false, false);
    acc[1][g] = __builtin_amdgcn_wmma_f32_16x16x32_bf16(
        false, a[1].v, false, bb[i % 3].v, (short)0, acc[1][g], false, false);
  }
}

// acc = bias + x @ Wih^T for tiny input dim (4): scalar FMA injection into
// the C/D fragment layout (lane: N = lane&15; M = mt*16 + (lane>>4)*8 + r).
__device__ __forceinline__ void init_small(f32x8 (&acc)[2][4],
                                           const GLB float* __restrict__ b,
                                           const GLB float* __restrict__ Wih,
                                           const float* xbase, int xstride,
                                           int lane, int hb) {
  const int nn = lane & 15, hs = lane >> 4;
  float bv[4], wv[4][4];
#pragma unroll
  for (int g = 0; g < 4; ++g) {
    const int col = g * 256 + hb * 16 + nn;
    bv[g] = b[col];
    const GLB float* w = Wih + col * 4;
    wv[g][0] = w[0]; wv[g][1] = w[1]; wv[g][2] = w[2]; wv[g][3] = w[3];
  }
#pragma unroll
  for (int mt = 0; mt < 2; ++mt) {
#pragma unroll
    for (int r = 0; r < 8; ++r) {
      const float* xr = xbase + (mt * 16 + hs * 8 + r) * xstride;
      const float x0 = xr[0], x1 = xr[1], x2 = xr[2], x3 = xr[3];
#pragma unroll
      for (int g = 0; g < 4; ++g)
        acc[mt][g][r] = bv[g] + wv[g][0] * x0 + wv[g][1] * x1
                              + wv[g][2] * x2 + wv[g][3] * x3;
    }
  }
}

__device__ __forceinline__ void init_bias(f32x8 (&acc)[2][4],
                                          const GLB float* __restrict__ b,
                                          int lane, int hb) {
  const int nn = lane & 15;
#pragma unroll
  for (int g = 0; g < 4; ++g) {
    const float bv = b[g * 256 + hb * 16 + nn];
    acc[0][g] = vsplat(bv);
    acc[1][g] = vsplat(bv);
  }
}

// Gates (i,f,g,o) -> c,h update. c state lives in LDS at cS[slot*NTHREADS]
// (per-lane private slots, bank-conflict-free b32); h' written to LDS bf16
// row-major for the next step's A-operand.
__device__ __forceinline__ void lstm_update(const f32x8 (&acc)[2][4],
                                            float* cS,
                                            __bf16* hA, int lane, int hb) {
  const int nn = lane & 15, hs = lane >> 4;
  const int j = hb * 16 + nn;
#pragma unroll
  for (int mt = 0; mt < 2; ++mt) {
#pragma unroll
    for (int r = 0; r < 8; ++r) {
      const int slot = mt * 8 + r;
      const float ig = sigf(acc[mt][0][r]);
      const float fg = sigf(acc[mt][1][r]);
      const float gg = tanhfast(acc[mt][2][r]);
      const float og = sigf(acc[mt][3][r]);
      const float cn = fg * cS[slot * NTHREADS] + ig * gg;
      cS[slot * NTHREADS] = cn;
      const float hn = og * tanhfast(cn);
      hA[(mt * 16 + hs * 8 + r) * HDIM + j] = f2bf(hn);
    }
  }
}

// MLP layer 1: r = relu(h1 @ Wp1^T + bp1), N=128. Waves 0..7 -> M-tile 0,
// waves 8..15 -> M-tile 1; each wave owns one 16-col N-tile.
__device__ __forceinline__ void mlp_layer1(const __bf16* __restrict__ hA1,
                                           const GLB __bf16* __restrict__ Wp1b,
                                           const GLB float* __restrict__ bp1,
                                           __bf16* rbuf, int lane, int wave) {
  const int nt = wave & 7, mt = wave >> 3;
  const int nn = lane & 15, hs = lane >> 4;
  f32x8 m = vsplat(bp1[nt * 16 + nn]);
#pragma unroll
  for (int kb = 0; kb < 8; ++kb) {
    Frag16 a;
    const __bf16* ab = hA1 + (mt * 16 + nn) * HDIM + kb * 32 + hs * 8;
    a.p[0] = *(const bf16x8*)(ab);
    a.p[1] = *(const bf16x8*)(ab + 16);
    Frag16 b;
    const GLB __bf16* wb = Wp1b + (nt * 16 + nn) * HDIM + kb * 32 + hs * 8;
    b.p[0] = *(const GLB bf16x8*)(wb);
    b.p[1] = *(const GLB bf16x8*)(wb + 16);
    m = __builtin_amdgcn_wmma_f32_16x16x32_bf16(
        false, a.v, false, b.v, (short)0, m, false, false);
  }
#pragma unroll
  for (int r = 0; r < 8; ++r) {
    const int row = mt * 16 + hs * 8 + r;
    float v = m[r];
    v = v > 0.0f ? v : 0.0f;
    rbuf[row * 128 + nt * 16 + nn] = f2bf(v);
  }
}

__global__ void cvt_bf16(const float* __restrict__ src,
                         __bf16* __restrict__ dst, int n) {
  const int i = blockIdx.x * blockDim.x + threadIdx.x;
  if (i < n) dst[i] = f2bf(src[i]);
}

__global__ void __launch_bounds__(NTHREADS)
lstm_seq2seq(const float* __restrict__ x,
             const float* __restrict__ enc_Wih0, const float* __restrict__ enc_b0,
             const float* __restrict__ enc_b1,
             const float* __restrict__ dec_Wih0, const float* __restrict__ dec_b0,
             const float* __restrict__ dec_b1,
             const float* __restrict__ bp1, const float* __restrict__ Wp2,
             const float* __restrict__ bp2,
             const __bf16* __restrict__ eWhh0, const __bf16* __restrict__ eWih1,
             const __bf16* __restrict__ eWhh1,
             const __bf16* __restrict__ dWhh0, const __bf16* __restrict__ dWih1,
             const __bf16* __restrict__ dWhh1,
             const __bf16* __restrict__ Wp1b,
             float* __restrict__ out) {
  __shared__ __attribute__((aligned(16))) __bf16 hA0[MROWS * HDIM];  // 16 KB
  __shared__ __attribute__((aligned(16))) __bf16 hA1[MROWS * HDIM];  // 16 KB
  __shared__ __attribute__((aligned(16))) __bf16 rbuf[MROWS * 128];  //  8 KB
  __shared__ float cS0[16 * NTHREADS];                               // 32 KB
  __shared__ float cS1[16 * NTHREADS];                               // 32 KB
  __shared__ float pred[MROWS * 4];

  const int tid = threadIdx.x;
  const int wave = tid >> 5, lane = tid & 31;
  const int hb = wave;                       // hidden block 0..15 (16 units)
  const int batch0 = blockIdx.x * MROWS;

  for (int i = tid; i < MROWS * HDIM; i += NTHREADS) {
    hA0[i] = f2bf(0.0f);
    hA1[i] = f2bf(0.0f);
  }
  for (int i = tid; i < 16 * NTHREADS; i += NTHREADS) {
    cS0[i] = 0.0f;
    cS1[i] = 0.0f;
  }
  float* c0p = &cS0[tid];   // per-lane private c slots, stride NTHREADS
  float* c1p = &cS1[tid];
  f32x8 acc[2][4];
  __syncthreads();

  // -------------------------- encoder (64 steps) --------------------------
#pragma unroll 1
  for (int t = 0; t < SEQ; ++t) {
    // Re-launder all loop-invariant pointers (blocks cross-step hoisting),
    // then pin them to the global address space for global_load_b128.
    const GLB __bf16* w_hh0 = (const GLB __bf16*)opaque(eWhh0);
    const GLB __bf16* w_ih1 = (const GLB __bf16*)opaque(eWih1);
    const GLB __bf16* w_hh1 = (const GLB __bf16*)opaque(eWhh1);
    const GLB float*  b0p   = (const GLB float*)opaque(enc_b0);
    const GLB float*  b1p   = (const GLB float*)opaque(enc_b1);
    const GLB float*  wih0p = (const GLB float*)opaque(enc_Wih0);

    init_small(acc, b0p, wih0p, x + (size_t)batch0 * XROW + t * 4, XROW, lane, hb);
    gemm_acc(acc, hA0, w_hh0, lane, hb);
    __syncthreads();                         // all reads of hA0 done
    lstm_update(acc, c0p, hA0, lane, hb);    // overwrite h0 in LDS
    __syncthreads();

    init_bias(acc, b1p, lane, hb);
    gemm_acc(acc, hA0, w_ih1, lane, hb);
    gemm_acc(acc, hA1, w_hh1, lane, hb);
    __syncthreads();
    lstm_update(acc, c1p, hA1, lane, hb);
    __syncthreads();
  }

  // decoder input init: last x timestep
  if (tid < MROWS * 4)
    pred[tid] = x[(size_t)(batch0 + (tid >> 2)) * XROW + (SEQ - 1) * 4 + (tid & 3)];
  __syncthreads();

  // -------------------------- decoder (32 steps) --------------------------
#pragma unroll 1
  for (int t = 0; t < TLEN; ++t) {
    const GLB __bf16* w_hh0 = (const GLB __bf16*)opaque(dWhh0);
    const GLB __bf16* w_ih1 = (const GLB __bf16*)opaque(dWih1);
    const GLB __bf16* w_hh1 = (const GLB __bf16*)opaque(dWhh1);
    const GLB __bf16* w_p1  = (const GLB __bf16*)opaque(Wp1b);
    const GLB float*  b0p   = (const GLB float*)opaque(dec_b0);
    const GLB float*  b1p   = (const GLB float*)opaque(dec_b1);
    const GLB float*  wih0p = (const GLB float*)opaque(dec_Wih0);
    const GLB float*  bp1p  = (const GLB float*)opaque(bp1);
    const GLB float*  wp2p  = (const GLB float*)opaque(Wp2);
    const GLB float*  bp2p  = (const GLB float*)opaque(bp2);

    init_small(acc, b0p, wih0p, pred, 4, lane, hb);
    gemm_acc(acc, hA0, w_hh0, lane, hb);
    __syncthreads();
    lstm_update(acc, c0p, hA0, lane, hb);
    __syncthreads();

    init_bias(acc, b1p, lane, hb);
    gemm_acc(acc, hA0, w_ih1, lane, hb);
    gemm_acc(acc, hA1, w_hh1, lane, hb);
    __syncthreads();
    lstm_update(acc, c1p, hA1, lane, hb);
    __syncthreads();

    // projection MLP (autoregressive feedback)
    mlp_layer1(hA1, w_p1, bp1p, rbuf, lane, wave);
    __syncthreads();
    if (tid < MROWS * 4) {
      const int m = tid >> 2, n = tid & 3;
      float s = bp2p[n];
#pragma unroll 8
      for (int k = 0; k < 128; ++k)
        s += bf2f(rbuf[m * 128 + k]) * wp2p[n * 128 + k];
      const float p = sigf(s);
      pred[m * 4 + n] = p;
      out[((size_t)(batch0 + m) * TLEN + t) * 4 + n] = p;
    }
    __syncthreads();
  }
}

extern "C" void kernel_launch(void* const* d_in, const int* in_sizes, int n_in,
                              void* d_out, int out_size, void* d_ws, size_t ws_size,
                              hipStream_t stream) {
  (void)in_sizes; (void)n_in; (void)out_size; (void)ws_size;

  const float* x      = (const float*)d_in[0];
  const float* eWih0f = (const float*)d_in[1];
  const float* eWhh0f = (const float*)d_in[2];
  const float* eb0    = (const float*)d_in[3];
  const float* eWih1f = (const float*)d_in[4];
  const float* eWhh1f = (const float*)d_in[5];
  const float* eb1    = (const float*)d_in[6];
  const float* dWih0f = (const float*)d_in[7];
  const float* dWhh0f = (const float*)d_in[8];
  const float* db0    = (const float*)d_in[9];
  const float* dWih1f = (const float*)d_in[10];
  const float* dWhh1f = (const float*)d_in[11];
  const float* db1    = (const float*)d_in[12];
  const float* Wp1f   = (const float*)d_in[13];
  const float* bp1    = (const float*)d_in[14];
  const float* Wp2    = (const float*)d_in[15];
  const float* bp2    = (const float*)d_in[16];

  // bf16 weight cache in workspace (~3.1 MB)
  __bf16* ws = (__bf16*)d_ws;
  const int WSZ = 1024 * 256;
  __bf16* eWhh0 = ws + 0 * (size_t)WSZ;
  __bf16* eWih1 = ws + 1 * (size_t)WSZ;
  __bf16* eWhh1 = ws + 2 * (size_t)WSZ;
  __bf16* dWhh0 = ws + 3 * (size_t)WSZ;
  __bf16* dWih1 = ws + 4 * (size_t)WSZ;
  __bf16* dWhh1 = ws + 5 * (size_t)WSZ;
  __bf16* Wp1b  = ws + 6 * (size_t)WSZ;

  cvt_bf16<<<WSZ / 256, 256, 0, stream>>>(eWhh0f, eWhh0, WSZ);
  cvt_bf16<<<WSZ / 256, 256, 0, stream>>>(eWih1f, eWih1, WSZ);
  cvt_bf16<<<WSZ / 256, 256, 0, stream>>>(eWhh1f, eWhh1, WSZ);
  cvt_bf16<<<WSZ / 256, 256, 0, stream>>>(dWhh0f, dWhh0, WSZ);
  cvt_bf16<<<WSZ / 256, 256, 0, stream>>>(dWih1f, dWih1, WSZ);
  cvt_bf16<<<WSZ / 256, 256, 0, stream>>>(dWhh1f, dWhh1, WSZ);
  cvt_bf16<<<(128 * 256) / 256, 256, 0, stream>>>(Wp1f, Wp1b, 128 * 256);

  lstm_seq2seq<<<4096 / MROWS, NTHREADS, 0, stream>>>(
      x, eWih0f, eb0, eb1, dWih0f, db0, db1, bp1, Wp2, bp2,
      eWhh0, eWih1, eWhh1, dWhh0, dWih1, dWhh1, Wp1b, (float*)d_out);
}
